// DoubleAttentionPairBlock_17995912970894
// MI455X (gfx1250) — matrix operand order repaired and checked by reading, not compiled, not run on hardware
//
#include <hip/hip_runtime.h>
#include <hip/hip_bf16.h>
#include <math.h>

typedef _Float16 h16;
typedef __attribute__((ext_vector_type(16))) _Float16 v16h;
typedef __attribute__((ext_vector_type(4)))  _Float16 v4h;
typedef __attribute__((ext_vector_type(8)))  float    v8f;

#define BB 2
#define NT 384
#define NNE (NT*NT)     // 147456 edges per batch
#define CC 128
#define DD 32
#define HHID 128

// ---- WMMA fragment helpers (CDNA5 16x16x32 f16 layouts, wave32) ----------
// A (16-bit 16x32, MxK): lane&15 = M; lanes 0-15 kb=0, lanes 16-31 kb=8.
// elements 0..7 = K[k0+kb .. k0+kb+7], elements 8..15 = K[k0+16+kb .. +7].
__device__ __forceinline__ v16h load_fragA(const h16* lds, int stride, int m0,
                                           int k0, int lane) {
  const int m  = m0 + (lane & 15);
  const int kb = (lane & 16) ? 8 : 0;
  const h16* p = lds + m * stride + k0 + kb;
  v16h a;
#pragma unroll
  for (int j = 0; j < 8; ++j) a[j] = p[j];
#pragma unroll
  for (int j = 0; j < 8; ++j) a[8 + j] = p[16 + j];
  return a;
}

// B (16-bit 32x16, KxN) from transposed weights WT[n][k]:
// lane&15 = N; lanes 0-15 kb=0, lanes 16-31 kb=8; elements 0..7 = K[k0+kb..+7], 8..15 = K[k0+16+kb..+7].
__device__ __forceinline__ v16h load_fragB(const h16* lds, int stride, int n0,
                                           int k0, int lane) {
  const int n  = n0 + (lane & 15);
  const int kb = (lane & 16) ? 8 : 0;
  const h16* p = lds + n * stride + k0 + kb;
  v16h b;
#pragma unroll
  for (int j = 0; j < 8; ++j) b[j] = p[j];
#pragma unroll
  for (int j = 0; j < 8; ++j) b[8 + j] = p[16 + j];
  return b;
}

__device__ __forceinline__ v8f wmma16(v16h a, v16h b, v8f c) {
  return __builtin_amdgcn_wmma_f32_16x16x32_f16(false, a, false, b,
                                                (short)0, c, false, false);
}

__device__ __forceinline__ v4h cvt4(float4 v) {
  v4h h = {(h16)v.x, (h16)v.y, (h16)v.z, (h16)v.w};
  return h;
}

// ---- kernel 0: zero the accumulators in workspace -------------------------
__global__ void k_zero(float* __restrict__ p, int n) {
  int i = blockIdx.x * blockDim.x + threadIdx.x;
  if (i < n) p[i] = 0.f;
}

// ---- kernel 1: fused fP/gP GEMMs + masked exp + global context accum ------
// grid (NNE/64, B), 256 threads (8 waves). Tile = 64 edge rows.
__global__ __launch_bounds__(256) void k_context(
    const float* __restrict__ edge, const int* __restrict__ mask,
    const float* __restrict__ Wg, const float* __restrict__ bg,
    const float* __restrict__ Wv, const float* __restrict__ bv,
    float* __restrict__ Sg, float* __restrict__ Vp)
{
  __shared__ __align__(16) h16 Pt[64 * 136];    // P tile, f16
  __shared__ __align__(16) h16 WgT[32 * 136];   // Wg^T [d][k]
  __shared__ __align__(16) h16 WvT[32 * 136];   // Wv^T [d][k]
  __shared__ __align__(16) h16 ET[32 * 72];     // e^T  [d][m]
  __shared__ __align__(16) h16 GT[32 * 72];     // gP^T [e][m]
  __shared__ float mfl[64];
  __shared__ float Sl[32];

  const int tid  = threadIdx.x;
  const int lane = tid & 31;
  const int wid  = tid >> 5;
  const int b    = blockIdx.y;
  const size_t rowbase = (size_t)b * NNE + (size_t)blockIdx.x * 64;

  { // cooperative loads (b128 global, packed f16 LDS stores)
    const float4* src4 = (const float4*)(edge + rowbase * CC);
#pragma unroll
    for (int i = tid; i < (64 * CC) / 4; i += 256) {
      const float4 v = src4[i];
      const int r = i >> 5;            // (i*4) >> 7
      const int c = (i & 31) * 4;      // (i*4) & 127
      *(v4h*)&Pt[r * 136 + c] = cvt4(v);
    }
    const float4* wg4 = (const float4*)Wg;
    const float4* wv4 = (const float4*)Wv;
#pragma unroll
    for (int i = tid; i < (CC * DD) / 4; i += 256) {
      const int k = i >> 3;            // (i*4) >> 5
      const int d = (i & 7) * 4;
      const float4 g = wg4[i];
      const float4 v = wv4[i];
      WgT[(d + 0) * 136 + k] = (h16)g.x;
      WgT[(d + 1) * 136 + k] = (h16)g.y;
      WgT[(d + 2) * 136 + k] = (h16)g.z;
      WgT[(d + 3) * 136 + k] = (h16)g.w;
      WvT[(d + 0) * 136 + k] = (h16)v.x;
      WvT[(d + 1) * 136 + k] = (h16)v.y;
      WvT[(d + 2) * 136 + k] = (h16)v.z;
      WvT[(d + 3) * 136 + k] = (h16)v.w;
    }
    if (tid < 64) mfl[tid] = (mask[rowbase + tid] != 0) ? 1.f : 0.f;
    if (tid < 32) Sl[tid] = 0.f;
  }
  __syncthreads();

  { // phase A: fP = P@Wg, gP = P@Wv ; e = mask*exp(fP)
    const int m0 = (wid >> 1) * 16;
    const int n0 = (wid & 1) * 16;
    v8f cF = {};
    v8f cG = {};
#pragma unroll
    for (int kt = 0; kt < 4; ++kt) {
      v16h a = load_fragA(Pt, 136, m0, kt * 32, lane);
      cF = wmma16(a, load_fragB(WgT, 136, n0, kt * 32, lane), cF);
      cG = wmma16(a, load_fragB(WvT, 136, n0, kt * 32, lane), cG);
    }
    const int d = n0 + (lane & 15);
    const float bgv = bg[d];
    const float bvv = bv[d];
    float ssum = 0.f;
#pragma unroll
    for (int r = 0; r < 8; ++r) {
      const int m = m0 + r + ((lane & 16) ? 8 : 0);
      const float e = mfl[m] * expf(cF[r] + bgv);
      const float g = cG[r] + bvv;
      ssum += e;
      ET[d * 72 + m] = (h16)e;
      GT[d * 72 + m] = (h16)g;
    }
    atomicAdd(&Sl[d], ssum);
  }
  __syncthreads();

  // phase B: Vp[d][e] += e^T @ gP  (32x32, K = 64 tile rows)
  if (wid < 4) {
    const int d0 = (wid >> 1) * 16;
    const int e0 = (wid & 1) * 16;
    v8f c = {};
#pragma unroll
    for (int kt = 0; kt < 2; ++kt)
      c = wmma16(load_fragA(ET, 72, d0, kt * 32, lane),
                 load_fragB(GT, 72, e0, kt * 32, lane), c);
    float* dst = Vp + b * (DD * DD);
#pragma unroll
    for (int r = 0; r < 8; ++r) {
      const int dd = d0 + r + ((lane & 16) ? 8 : 0);
      atomicAdd(dst + dd * DD + e0 + (lane & 15), c[r]);
    }
  } else if (wid == 4) {
    atomicAdd(&Sg[b * DD + lane], Sl[lane]);
  }
}

// ---- kernel 2: normalize context V and compute FiLM MLP (tiny) ------------
__global__ __launch_bounds__(256) void k_finalize(
    const float* __restrict__ Vp, const float* __restrict__ Sg,
    float* __restrict__ Vn, float* __restrict__ film,
    const float* __restrict__ r3t,
    const float* __restrict__ W1, const float* __restrict__ b1,
    const float* __restrict__ W2, const float* __restrict__ b2)
{
  __shared__ float h[HHID];
  const int tid = threadIdx.x;
  for (int i = tid; i < BB * DD * DD; i += 256) {
    const int b = i >> 10;
    const int d = (i >> 5) & 31;
    Vn[i] = Vp[i] / (Sg[b * DD + d] + 1e-8f);
  }
  for (int b = 0; b < BB; ++b) {
    __syncthreads();
    if (tid < HHID) {
      const float x = r3t[b] * W1[tid] + b1[tid];
      h[tid] = 0.5f * x * (1.f + erff(x * 0.70710678118654752f)); // exact GELU
    }
    __syncthreads();
    float acc = b2[tid];                       // tid indexes 2*CC = 256 outs
    for (int k = 0; k < HHID; ++k) acc += h[k] * W2[k * (2 * CC) + tid];
    film[b * (2 * CC) + tid] = acc;
  }
}

// ---- kernel 3: hP softmax, O = hP@V@Wo, FiLM, residual --------------------
// grid (NNE/128, B), 256 threads (8 waves). Tile = 128 edge rows.
__global__ __launch_bounds__(256) void k_output(
    const float* __restrict__ edge,
    const float* __restrict__ Wd, const float* __restrict__ bd,
    const float* __restrict__ Wo, const float* __restrict__ bo,
    const float* __restrict__ Vn, const float* __restrict__ film,
    float* __restrict__ out)
{
  __shared__ __align__(16) unsigned char smem[43520];
  __shared__ float fS[CC], fH[CC];
  h16* Pt  = (h16*)smem;                 // phase 1: 128x136 (34816 B)
  h16* WdT = (h16*)(smem + 34816);       // phase 1: 32x136  (8704 B)
  h16* Hs  = (h16*)smem;                 // phase 2: 128x40  (10240 B)
  h16* Vt  = (h16*)(smem + 10240);       // phase 2: 32x40   (2560 B)
  h16* Os  = (h16*)(smem + 12800);       // phase 2/3: 128x40 (10240 B)
  h16* WoT = (h16*)(smem + 23040);       // phase 3: 128x40  (10240 B)

  const int tid  = threadIdx.x;
  const int lane = tid & 31;
  const int wid  = tid >> 5;
  const int b    = blockIdx.y;
  const size_t rowbase = (size_t)b * NNE + (size_t)blockIdx.x * 128;

  {
    const float4* src4 = (const float4*)(edge + rowbase * CC);
#pragma unroll
    for (int i = tid; i < (128 * CC) / 4; i += 256) {
      const float4 v = src4[i];
      const int r = i >> 5;
      const int c = (i & 31) * 4;
      *(v4h*)&Pt[r * 136 + c] = cvt4(v);
    }
    const float4* wd4 = (const float4*)Wd;
#pragma unroll
    for (int i = tid; i < (CC * DD) / 4; i += 256) {
      const int k = i >> 3;
      const int d = (i & 7) * 4;
      const float4 v = wd4[i];
      WdT[(d + 0) * 136 + k] = (h16)v.x;
      WdT[(d + 1) * 136 + k] = (h16)v.y;
      WdT[(d + 2) * 136 + k] = (h16)v.z;
      WdT[(d + 3) * 136 + k] = (h16)v.w;
    }
    if (tid < CC) {
      fS[tid] = film[b * 256 + tid];
      fH[tid] = film[b * 256 + 128 + tid];
    }
  }
  __syncthreads();

  const int m0  = wid * 16;
  const int dlo = lane & 15;
  float hA[8], hB[8];
  { // phase 1: logits = P@Wd + bd, channel softmax in registers
    v8f c0 = {}, c1 = {};
#pragma unroll
    for (int kt = 0; kt < 4; ++kt) {
      v16h a = load_fragA(Pt, 136, m0, kt * 32, lane);
      c0 = wmma16(a, load_fragB(WdT, 136, 0,  kt * 32, lane), c0);
      c1 = wmma16(a, load_fragB(WdT, 136, 16, kt * 32, lane), c1);
    }
    const float bd0 = bd[dlo], bd1 = bd[16 + dlo];
#pragma unroll
    for (int r = 0; r < 8; ++r) {
      const float x0 = c0[r] + bd0;
      const float x1 = c1[r] + bd1;
      float mx = fmaxf(x0, x1);         // row values live in one 16-lane half
#pragma unroll
      for (int off = 1; off < 16; off <<= 1)
        mx = fmaxf(mx, __shfl_xor(mx, off, 32));
      const float e0 = expf(x0 - mx), e1 = expf(x1 - mx);
      float s = e0 + e1;
#pragma unroll
      for (int off = 1; off < 16; off <<= 1) s += __shfl_xor(s, off, 32);
      const float inv = 1.f / s;
      hA[r] = e0 * inv;
      hB[r] = e1 * inv;
    }
  }
  __syncthreads();                       // Pt/WdT dead; overlay region live

  { // store hP tile, load V^T and Wo^T into overlay
#pragma unroll
    for (int r = 0; r < 8; ++r) {
      const int m = m0 + r + ((lane & 16) ? 8 : 0);
      Hs[m * 40 + dlo]      = (h16)hA[r];
      Hs[m * 40 + 16 + dlo] = (h16)hB[r];
    }
    const float4* vn4 = (const float4*)(Vn + b * (DD * DD));
#pragma unroll
    for (int i = tid; i < (DD * DD) / 4; i += 256) {   // Vt[e][d] = V[d][e]
      const int d = i >> 3;
      const int e = (i & 7) * 4;
      const float4 v = vn4[i];
      Vt[(e + 0) * 40 + d] = (h16)v.x;
      Vt[(e + 1) * 40 + d] = (h16)v.y;
      Vt[(e + 2) * 40 + d] = (h16)v.z;
      Vt[(e + 3) * 40 + d] = (h16)v.w;
    }
    const float4* wo4 = (const float4*)Wo;
#pragma unroll
    for (int i = tid; i < (CC * DD) / 4; i += 256) {   // WoT[c][d] = Wo[d][c]
      const int d = i >> 5;
      const int c = (i & 31) * 4;
      const float4 v = wo4[i];
      WoT[(c + 0) * 40 + d] = (h16)v.x;
      WoT[(c + 1) * 40 + d] = (h16)v.y;
      WoT[(c + 2) * 40 + d] = (h16)v.z;
      WoT[(c + 3) * 40 + d] = (h16)v.w;
    }
  }
  __syncthreads();

  { // phase 2: O = hP @ V  (K = 32, single WMMA per tile)
    v8f cz = {};
    v16h a  = load_fragA(Hs, 40, m0, 0, lane);
    v8f c0 = wmma16(a, load_fragB(Vt, 40, 0,  0, lane), cz);
    v8f c1 = wmma16(a, load_fragB(Vt, 40, 16, 0, lane), cz);
#pragma unroll
    for (int r = 0; r < 8; ++r) {
      const int m = m0 + r + ((lane & 16) ? 8 : 0);
      Os[m * 40 + dlo]      = (h16)c0[r];
      Os[m * 40 + 16 + dlo] = (h16)c1[r];
    }
  }
  __syncthreads();

  { // phase 3: O2 = O @ Wo + bo -> FiLM -> residual -> store
    v8f cz = {};
    v16h a = load_fragA(Os, 40, m0, 0, lane);        // reused across 8 tiles
#pragma unroll
    for (int nt = 0; nt < 8; ++nt) {
      v8f c = wmma16(a, load_fragB(WoT, 40, nt * 16, 0, lane), cz);
      const int cc   = nt * 16 + dlo;
      const float bov = bo[cc];
      const float sc  = 1.f + fS[cc];
      const float sh  = fH[cc];
#pragma unroll
      for (int r = 0; r < 8; ++r) {
        const int m = m0 + r + ((lane & 16) ? 8 : 0);
        const size_t gi = (rowbase + m) * CC + cc;
        out[gi] = edge[gi] + (c[r] + bov) * sc + sh;
      }
    }
  }
}

// ---- host-side launch ------------------------------------------------------
extern "C" void kernel_launch(void* const* d_in, const int* in_sizes, int n_in,
                              void* d_out, int out_size, void* d_ws, size_t ws_size,
                              hipStream_t stream) {
  (void)in_sizes; (void)n_in; (void)out_size; (void)ws_size;
  const float* edge = (const float*)d_in[0];
  const int*   mask = (const int*)  d_in[1];
  const float* r3t  = (const float*)d_in[2];
  const float* Wg   = (const float*)d_in[3];
  const float* bg   = (const float*)d_in[4];
  const float* Wv   = (const float*)d_in[5];
  const float* bv   = (const float*)d_in[6];
  const float* Wd   = (const float*)d_in[7];
  const float* bd   = (const float*)d_in[8];
  const float* Wo   = (const float*)d_in[9];
  const float* bo   = (const float*)d_in[10];
  const float* W1   = (const float*)d_in[11];
  const float* b1   = (const float*)d_in[12];
  const float* W2   = (const float*)d_in[13];
  const float* b2   = (const float*)d_in[14];
  float* out = (float*)d_out;

  float* wsf  = (float*)d_ws;
  float* Sg   = wsf;          // [B,32]   column sums
  float* Vp   = wsf + 64;     // [B,32,32] unnormalized context
  float* Vn   = wsf + 2112;   // [B,32,32] normalized context
  float* film = wsf + 4160;   // [B,256]  scale|shift

  hipLaunchKernelGGL(k_zero, dim3(9), dim3(256), 0, stream, wsf, 2112);
  hipLaunchKernelGGL(k_context, dim3(NNE / 64, BB), dim3(256), 0, stream,
                     edge, mask, Wg, bg, Wv, bv, Sg, Vp);
  hipLaunchKernelGGL(k_finalize, dim3(1), dim3(256), 0, stream,
                     Vp, Sg, Vn, film, r3t, W1, b1, W2, b2);
  hipLaunchKernelGGL(k_output, dim3(NNE / 128, BB), dim3(256), 0, stream,
                     edge, Wd, bd, Wo, bo, Vn, film, out);
}
